// ImageProjectionLayer_14302241096002
// MI455X (gfx1250) — 8-bit matrix operand order repaired and checked by reading, compiled, not run on hardware
//
#include <hip/hip_runtime.h>
#include <cstdint>

typedef int v8i __attribute__((ext_vector_type(8)));
typedef int v4i __attribute__((ext_vector_type(4)));

#define D_IN  1024
#define D_OUT 2560
#define KSTEP 64

#if defined(__HIP_DEVICE_COMPILE__) && \
    __has_builtin(__builtin_amdgcn_global_load_async_to_lds_b64)
#define USE_ASYNC_LDS 1
typedef int v2i_g __attribute__((vector_size(8)));
typedef __attribute__((address_space(1))) v2i_g* gptr_b64;
typedef __attribute__((address_space(3))) v2i_g* lptr_b64;
#else
#define USE_ASYNC_LDS 0
#endif

__device__ __forceinline__ int clamp127(int v) {
    return v < -127 ? -127 : (v > 127 ? 127 : v);
}

__global__ __launch_bounds__(256)
void quant_rows(const float* __restrict__ x, uint32_t* __restrict__ xq,
                float* __restrict__ sx) {
    const int row = blockIdx.x;
    const int tid = threadIdx.x;
    const float4 v =
        reinterpret_cast<const float4*>(x + (size_t)row * D_IN)[tid];
    float m = fmaxf(fmaxf(fabsf(v.x), fabsf(v.y)),
                    fmaxf(fabsf(v.z), fabsf(v.w)));
#pragma unroll
    for (int off = 16; off > 0; off >>= 1)
        m = fmaxf(m, __shfl_xor(m, off, 32));
    __shared__ float red[8];
    if ((tid & 31) == 0) red[tid >> 5] = m;
    __syncthreads();
    float mm = fmaxf(fmaxf(fmaxf(red[0], red[1]), fmaxf(red[2], red[3])),
                     fmaxf(fmaxf(red[4], red[5]), fmaxf(red[6], red[7])));
    mm = fmaxf(mm, 1e-8f);
    const float inv = 127.0f / mm;

    const int q0 = clamp127((int)rintf(v.x * inv));
    const int q1 = clamp127((int)rintf(v.y * inv));
    const int q2 = clamp127((int)rintf(v.z * inv));
    const int q3 = clamp127((int)rintf(v.w * inv));
    const uint32_t p = (uint32_t)(q0 & 0xff) | ((uint32_t)(q1 & 0xff) << 8) |
                       ((uint32_t)(q2 & 0xff) << 16) |
                       ((uint32_t)(q3 & 0xff) << 24);
    xq[(size_t)row * (D_IN / 4) + tid] = p;
    if (tid == 0) sx[row] = mm / 127.0f;
}

__global__ __launch_bounds__(256)
void pack_w(const int* __restrict__ w, uint32_t* __restrict__ wq, int n4) {
    const int i = blockIdx.x * 256 + threadIdx.x;
    if (i >= n4) return;
    const int4 v = reinterpret_cast<const int4*>(w)[i];
    wq[i] = (uint32_t)(v.x & 0xff) | ((uint32_t)(v.y & 0xff) << 8) |
            ((uint32_t)(v.z & 0xff) << 16) | ((uint32_t)(v.w & 0xff) << 24);
}

__global__ __launch_bounds__(256)
void gemm_wmma_iu8(const uint8_t* __restrict__ xq,
                   const uint8_t* __restrict__ wq,
                   const float* __restrict__ sx, const float* __restrict__ sw,
                   const float* __restrict__ bias, float* __restrict__ out) {
    __shared__ uint8_t smA[32 * D_IN];

    const int tid = threadIdx.x;
    const int lane = tid & 31;
    const int wave = tid >> 5;
    const int m0 = blockIdx.y * 32;
    const int n0 = blockIdx.x * 256 + wave * 32;

    const int l15 = lane & 15;
    const int hi = lane >> 4;

    {
        const uint8_t* gbase = xq + (size_t)m0 * D_IN;
#pragma unroll
        for (int t = 0; t < 16; ++t) {
            const int c = t * 256 + tid;
            const int row = c >> 7;
            const int kb = (c & 127) << 3;
            const int rb = row >> 4;
            const int ks = kb >> 6;
            const int rem = kb & 63;
            const int j = ((rem >> 5) << 1) | ((rem >> 3) & 1);
            const int h8 = (rem >> 4) & 1;
            const int flane = (row & 15) + (h8 << 4);
            const int loff = (rb * 16 + ks) * 1024 + (j >> 1) * 512 +
                             flane * 16 + (j & 1) * 8;
            const int goff = row * D_IN + kb;
#if USE_ASYNC_LDS
            __builtin_amdgcn_global_load_async_to_lds_b64(
                (gptr_b64)(gbase + goff), (lptr_b64)(smA + loff), 0, 0);
#else
            *reinterpret_cast<uint64_t*>(smA + loff) =
                *reinterpret_cast<const uint64_t*>(gbase + goff);
#endif
        }
    }
#if USE_ASYNC_LDS
#if __has_builtin(__builtin_amdgcn_s_wait_asynccnt)
    __builtin_amdgcn_s_wait_asynccnt(0);
#else
    asm volatile("s_wait_asynccnt 0x0" ::: "memory");
#endif
#endif
    __syncthreads();

    const uint8_t* abase = smA + lane * 16;
    const uint8_t* bcol0 = wq + (size_t)(n0 + l15) * D_IN + hi * 16;
    const uint8_t* bcol1 = wq + (size_t)(n0 + 16 + l15) * D_IN + hi * 16;

    v8i acc00 = {}, acc01 = {}, acc10 = {}, acc11 = {};

#pragma unroll
    for (int k = 0; k < D_IN; k += KSTEP) {
        const int koff = k * 16;
        const v4i a0lo = *reinterpret_cast<const v4i*>(abase + koff);
        const v4i a0hi = *reinterpret_cast<const v4i*>(abase + koff + 512);
        const v4i a1lo = *reinterpret_cast<const v4i*>(abase + 16384 + koff);
        const v4i a1hi =
            *reinterpret_cast<const v4i*>(abase + 16384 + koff + 512);
        const v8i a0 = __builtin_shufflevector(a0lo, a0hi, 0, 1, 2, 3, 4, 5, 6, 7);
        const v8i a1 = __builtin_shufflevector(a1lo, a1hi, 0, 1, 2, 3, 4, 5, 6, 7);

        const v4i b0lo = *reinterpret_cast<const v4i*>(bcol0 + k);
        const v4i b0hi = *reinterpret_cast<const v4i*>(bcol0 + k + 32);
        const v4i b1lo = *reinterpret_cast<const v4i*>(bcol1 + k);
        const v4i b1hi = *reinterpret_cast<const v4i*>(bcol1 + k + 32);
        const v8i b0 = __builtin_shufflevector(b0lo, b0hi, 0, 1, 2, 3, 4, 5, 6, 7);
        const v8i b1 = __builtin_shufflevector(b1lo, b1hi, 0, 1, 2, 3, 4, 5, 6, 7);

        acc00 = __builtin_amdgcn_wmma_i32_16x16x64_iu8(true, a0, true, b0,
                                                       acc00, false, false);
        acc01 = __builtin_amdgcn_wmma_i32_16x16x64_iu8(true, a0, true, b1,
                                                       acc01, false, false);
        acc10 = __builtin_amdgcn_wmma_i32_16x16x64_iu8(true, a1, true, b0,
                                                       acc10, false, false);
        acc11 = __builtin_amdgcn_wmma_i32_16x16x64_iu8(true, a1, true, b1,
                                                       acc11, false, false);
    }

    const int col0 = n0 + l15;
    const int col1 = n0 + 16 + l15;
    const float swc0 = sw[col0], bc0 = bias[col0];
    const float swc1 = sw[col1], bc1 = bias[col1];
#pragma unroll
    for (int r = 0; r < 8; ++r) {
        const int rowa = m0 + r + hi * 8;
        const int rowb = m0 + 16 + r + hi * 8;
        const float sa = sx[rowa];
        const float sb = sx[rowb];
        __builtin_nontemporal_store((float)acc00[r] * sa * swc0 + bc0,
                                    &out[(size_t)rowa * D_OUT + col0]);
        __builtin_nontemporal_store((float)acc01[r] * sa * swc1 + bc1,
                                    &out[(size_t)rowa * D_OUT + col1]);
        __builtin_nontemporal_store((float)acc10[r] * sb * swc0 + bc0,
                                    &out[(size_t)rowb * D_OUT + col0]);
        __builtin_nontemporal_store((float)acc11[r] * sb * swc1 + bc1,
                                    &out[(size_t)rowb * D_OUT + col1]);
    }
}

extern "C" void kernel_launch(void* const* d_in, const int* in_sizes, int n_in,
                              void* d_out, int out_size, void* d_ws,
                              size_t ws_size, hipStream_t stream) {
    const float* x = (const float*)d_in[0];
    const int* w = (const int*)d_in[1];
    const float* sw = (const float*)d_in[2];
    const float* bias = (const float*)d_in[3];
    float* out = (float*)d_out;

    const int M = in_sizes[0] / D_IN;

    uint8_t* ws = (uint8_t*)d_ws;
    size_t off = 0;
    uint32_t* xq = (uint32_t*)(ws + off);
    off += (size_t)M * D_IN;
    off = (off + 255) & ~(size_t)255;
    uint32_t* wq = (uint32_t*)(ws + off);
    off += (size_t)D_OUT * D_IN;
    off = (off + 255) & ~(size_t)255;
    float* sx = (float*)(ws + off);

    quant_rows<<<M, 256, 0, stream>>>(x, xq, sx);
    const int n4 = D_OUT * D_IN / 4;
    pack_w<<<(n4 + 255) / 256, 256, 0, stream>>>(w, wq, n4);
    gemm_wmma_iu8<<<dim3(D_OUT / 256, M / 32), 256, 0, stream>>>(
        (const uint8_t*)xq, (const uint8_t*)wq, sx, sw, bias, out);
}
